// ContextualAttentionNeck_60713657696348
// MI455X (gfx1250) — matrix operand order repaired and checked by reading, not compiled, not run on hardware
//
#include <hip/hip_runtime.h>
#include <hip/hip_bf16.h>

typedef __attribute__((ext_vector_type(16))) _Float16 v16h;
typedef __attribute__((ext_vector_type(8)))  _Float16 v8h_t;
typedef __attribute__((ext_vector_type(8)))  float    v8f;
typedef __attribute__((ext_vector_type(4)))  int      v4i;
typedef __attribute__((address_space(1))) v4i* g_v4i_p;  // global int4*
typedef __attribute__((address_space(3))) v4i* l_v4i_p;  // LDS int4*

#define B_   16
#define C_   128
#define H_   64
#define W_   64
#define HS   32      // downsampled spatial
#define L_   1024    // HS*HS
#define KC   1152    // C*9
#define NRAW 2048    // C*16

#define KSTEP 64     // K depth staged per barrier round (2 WMMA k-steps)
#define LROW  72     // padded LDS row pitch in halves (64 + 8 pad, 144 B)

#if defined(__has_builtin)
#if __has_builtin(__builtin_amdgcn_global_load_async_to_lds_b128)
#define HAVE_ASYNC_LDS 1
#endif
#endif
#ifndef HAVE_ASYNC_LDS
#define HAVE_ASYNC_LDS 0
#endif

__device__ __forceinline__ void ca_wait_async() {
#if HAVE_ASYNC_LDS
#if __has_builtin(__builtin_amdgcn_s_wait_asynccnt)
  __builtin_amdgcn_s_wait_asynccnt(0);
#else
  asm volatile("s_wait_asynccnt 0x0" ::: "memory");
#endif
#endif
}

// ---------------------------------------------------------------------------
// Prep: 3x3 patches of downsampled x (A = raw, B = L2-normalized), mask flag
// ---------------------------------------------------------------------------
__global__ void __launch_bounds__(128) ca_prep_cols(
    const float* __restrict__ x, const float* __restrict__ mask,
    _Float16* __restrict__ Acols, _Float16* __restrict__ Bcols,
    float* __restrict__ invalid) {
  int l = blockIdx.x, b = blockIdx.y, c = threadIdx.x;   // 128 threads = channels
  int ly = l >> 5, lx = l & 31;
  const float* xb = x + (size_t)(b * C_ + c) * (H_ * W_);
  float v[9];
  float ss = 0.f;
#pragma unroll
  for (int i = 0; i < 3; ++i) {
#pragma unroll
    for (int j = 0; j < 3; ++j) {
      int yy = ly + i - 1, xx = lx + j - 1;
      float val = 0.f;
      if (yy >= 0 && yy < HS && xx >= 0 && xx < HS)
        val = xb[(yy * 2) * W_ + (xx * 2)];
      v[i * 3 + j] = val;
      ss += val * val;
    }
  }
  __shared__ float red[128];
  red[c] = ss;
  __syncthreads();
  for (int s = 64; s > 0; s >>= 1) {
    if (c < s) red[c] += red[c + s];
    __syncthreads();
  }
  float rn = 1.f / fmaxf(sqrtf(red[0]), 1e-4f);
  size_t base = ((size_t)(b * L_ + l)) * KC + c * 9;
#pragma unroll
  for (int s2 = 0; s2 < 9; ++s2) {
    Acols[base + s2] = (_Float16)v[s2];
    Bcols[base + s2] = (_Float16)(v[s2] * rn);
  }
  if (c == 0) {
    const float* mb = mask + (size_t)b * (H_ * W_);
    float msum = 0.f;
    for (int i = 0; i < 3; ++i)
      for (int j = 0; j < 3; ++j) {
        int yy = ly + i - 1, xx = lx + j - 1;
        if (yy >= 0 && yy < HS && xx >= 0 && xx < HS)
          msum += mb[(yy * 2) * W_ + (xx * 2)];
      }
    invalid[b * L_ + l] = (msum > 0.f) ? 1.f : 0.f;
  }
}

// ---------------------------------------------------------------------------
// Prep: raw 4x4 stride-2 pad-1 patches, transposed layout rawT[b][n=c*16+ij][l]
// ---------------------------------------------------------------------------
__global__ void ca_prep_rawT(const float* __restrict__ x,
                             _Float16* __restrict__ rawT, int total) {
  int idx = blockIdx.x * blockDim.x + threadIdx.x;
  if (idx >= total) return;
  int l = idx & (L_ - 1);
  int n = (idx >> 10) & (NRAW - 1);
  int b = idx >> 21;
  int c = n >> 4, ij = n & 15, i = ij >> 2, j = ij & 3;
  int ly = l >> 5, lx = l & 31;
  int Y = 2 * ly + i - 1, X = 2 * lx + j - 1;
  float v = 0.f;
  if (Y >= 0 && Y < H_ && X >= 0 && X < W_)
    v = x[((size_t)(b * C_ + c) * H_ + Y) * W_ + X];
  rawT[idx] = (_Float16)v;
}

// ---------------------------------------------------------------------------
// WMMA fragment helpers (ISA 16-bit A 16x32 / B 32x16 / 32-bit C layouts)
// ---------------------------------------------------------------------------
__device__ __forceinline__ v16h ca_load_a(const _Float16* __restrict__ A,
                                          int lda, int kb, int lane) {
  const _Float16* p = A + (size_t)(lane & 15) * lda + kb + ((lane >> 4) << 3);
  v8h_t lo = *(const v8h_t*)(p);
  v8h_t hi = *(const v8h_t*)(p + 16);
  v16h a;
#pragma unroll
  for (int i = 0; i < 8; ++i) { a[i] = lo[i]; a[i + 8] = hi[i]; }
  return a;
}
__device__ __forceinline__ v16h ca_load_b_lds(const _Float16* lbase, int kk,
                                              int lane) {
  const _Float16* p = lbase + (lane & 15) * LROW + kk + ((lane >> 4) << 3);
  v8h_t lo = *(const v8h_t*)(p);
  v8h_t hi = *(const v8h_t*)(p + 16);
  v16h r;
#pragma unroll
  for (int i = 0; i < 8; ++i) { r[i] = lo[i]; r[8 + i] = hi[i]; }
  return r;
}

// Cooperative stage of the block's 64-row B strip, KSTEP deep, into LDS.
// 512 x 16B chunks, 2 per thread; async DMA to LDS when available.
__device__ __forceinline__ void ca_stage_b(const _Float16* __restrict__ Bb,
                                           int K, int kb,
                                           _Float16* __restrict__ lds) {
  int tid = threadIdx.x;
#pragma unroll
  for (int c = 0; c < 2; ++c) {
    int chunk = tid + c * 256;
    int n = chunk >> 3;
    int ko = (chunk & 7) << 3;   // halves within the KSTEP window
    const _Float16* src = Bb + (size_t)n * K + kb + ko;
    _Float16* dst = lds + n * LROW + ko;
#if HAVE_ASYNC_LDS
    _Float16* src_nc = const_cast<_Float16*>(src);
    __builtin_amdgcn_global_load_async_to_lds_b128(
        (g_v4i_p)src_nc, (l_v4i_p)dst, 0, 0);
#else
    *(v8h_t*)dst = *(const v8h_t*)src;
#endif
  }
}

// ---------------------------------------------------------------------------
// Batched NT GEMM with double-buffered LDS staging of the shared B strip.
// C[b][M,N] = A[b][M,K] * Bt[b][N,K]^T, block tile 128x64, wave tile 16x64.
// elu_mode=0: C[m*N+n] (row-major).  elu_mode=1: out[n*4096+m] = elu(acc+bias).
// ---------------------------------------------------------------------------
__global__ void __launch_bounds__(256) ca_gemm_staged(
    const _Float16* __restrict__ A, const _Float16* __restrict__ Bt,
    float* __restrict__ C, int N, int K,
    size_t sA, size_t sB, size_t sC,
    const float* __restrict__ bias, int elu_mode) {
  __shared__ _Float16 lbs[2][64 * LROW];
  int lane = threadIdx.x & 31, wave = threadIdx.x >> 5;
  int b = blockIdx.z;
  int m0 = blockIdx.y * 128 + wave * 16;
  int n0 = blockIdx.x * 64;
  const _Float16* Ab = A + (size_t)b * sA + (size_t)m0 * K;
  const _Float16* Bb = Bt + (size_t)b * sB + (size_t)n0 * K;
  v8f acc[4];
#pragma unroll
  for (int t = 0; t < 4; ++t)
#pragma unroll
    for (int r = 0; r < 8; ++r) acc[t][r] = 0.f;

  const int nst = K / KSTEP;
  ca_stage_b(Bb, K, 0, lbs[0]);
  ca_wait_async();
  __syncthreads();
  int buf = 0;
  for (int s = 0; s < nst; ++s) {
    if (s + 1 < nst) ca_stage_b(Bb, K, (s + 1) * KSTEP, lbs[buf ^ 1]);
    int kb = s * KSTEP;
#pragma unroll
    for (int kk = 0; kk < KSTEP; kk += 32) {
      v16h a = ca_load_a(Ab, K, kb + kk, lane);
#pragma unroll
      for (int t = 0; t < 4; ++t) {
        v16h bf = ca_load_b_lds(lbs[buf] + t * 16 * LROW, kk, lane);
        acc[t] = __builtin_amdgcn_wmma_f32_16x16x32_f16(
            false, a, false, bf, (short)0, acc[t], false, false);
      }
    }
    ca_wait_async();
    __syncthreads();
    buf ^= 1;
  }

  int nlane = lane & 15, half = lane >> 4;
  int mb = m0 + half * 8;
  if (elu_mode) {
#pragma unroll
    for (int t = 0; t < 4; ++t) {
      int n = n0 + 16 * t + nlane;
      float bv = bias[n];
#pragma unroll
      for (int r = 0; r < 8; ++r) {
        float v = acc[t][r] + bv;
        v = (v > 0.f) ? v : (__expf(v) - 1.f);
        C[(size_t)n * 4096 + (mb + r)] = v;   // [cout, pix] == NCHW flat
      }
    }
  } else {
    float* Cb = C + (size_t)b * sC;
#pragma unroll
    for (int t = 0; t < 4; ++t) {
      float* cp = Cb + (size_t)mb * N + n0 + 16 * t + nlane;
#pragma unroll
      for (int r = 0; r < 8; ++r) cp[(size_t)r * N] = acc[t][r];
    }
  }
}

// ---------------------------------------------------------------------------
// Fuse pass 1: diagonal 3-tap over flattened (pos,key) map
// ---------------------------------------------------------------------------
__global__ void ca_fuse1(const float* __restrict__ c, float* __restrict__ o,
                         int total) {
  int idx = blockIdx.x * blockDim.x + threadIdx.x;
  if (idx >= total) return;
  int l = idx & (L_ - 1);
  int p = (idx >> 10) & (L_ - 1);
  size_t base = (size_t)(idx >> 20) << 20;
  float s = c[idx];
  if (p > 0 && l > 0)           s += c[base + (size_t)(p - 1) * L_ + (l - 1)];
  if (p < L_ - 1 && l < L_ - 1) s += c[base + (size_t)(p + 1) * L_ + (l + 1)];
  o[idx] = s;
}

// ---------------------------------------------------------------------------
// Fuse pass 2 (in (x,y)/(xu,yu)-transposed flat layout) + mask + x10 scale
// ---------------------------------------------------------------------------
__global__ void ca_fuse2_mask(const float* __restrict__ t,
                              const float* __restrict__ invalid,
                              float* __restrict__ o, int total) {
  int idx = blockIdx.x * blockDim.x + threadIdx.x;
  if (idx >= total) return;
  int l = idx & (L_ - 1);
  int p = (idx >> 10) & (L_ - 1);
  int b = idx >> 20;
  int y = p >> 5, xx = p & 31, yu = l >> 5, xu = l & 31;
  int q = xx * 32 + y, r = xu * 32 + yu;
  size_t base = (size_t)b << 20;
  float s = 0.f;
#pragma unroll
  for (int dt = -1; dt <= 1; ++dt) {
    int q2 = q + dt, r2 = r + dt;
    if (q2 >= 0 && q2 < L_ && r2 >= 0 && r2 < L_) {
      int x2 = q2 >> 5, y2 = q2 & 31;
      int xu2 = r2 >> 5, yu2 = r2 & 31;
      s += t[base + (size_t)(y2 * 32 + x2) * L_ + (yu2 * 32 + xu2)];
    }
  }
  float inv = invalid[b * L_ + l];
  float v = s * (1.f - inv) + inv * (-10000.f);
  o[idx] = v * 10.f;   // fold SOFTMAX_SCALE
}

// ---------------------------------------------------------------------------
// Softmax over keys per (b,pos); emits f16 attn[b,p,l] (GEMM A) and f32
// attention map in reference layout [b,l,y,x] directly into d_out.
// ---------------------------------------------------------------------------
__global__ void __launch_bounds__(256) ca_softmax(
    const float* __restrict__ c, _Float16* __restrict__ attn_h,
    float* __restrict__ offset_out) {
  int row = blockIdx.x;              // b*1024 + p
  int b = row >> 10, p = row & (L_ - 1);
  const float* cr = c + (size_t)row * L_;
  int t = threadIdx.x;
  __shared__ float red[256];
  float m = -3.4e38f;
  for (int i = t; i < L_; i += 256) m = fmaxf(m, cr[i]);
  red[t] = m;
  __syncthreads();
  for (int s = 128; s > 0; s >>= 1) {
    if (t < s) red[t] = fmaxf(red[t], red[t + s]);
    __syncthreads();
  }
  m = red[0];
  __syncthreads();
  float e[4];
  float sum = 0.f;
#pragma unroll
  for (int k = 0; k < 4; ++k) { e[k] = __expf(cr[t + k * 256] - m); sum += e[k]; }
  red[t] = sum;
  __syncthreads();
  for (int s = 128; s > 0; s >>= 1) {
    if (t < s) red[t] += red[t + s];
    __syncthreads();
  }
  float rs = 1.f / red[0];
#pragma unroll
  for (int k = 0; k < 4; ++k) {
    int l = t + k * 256;
    float a = e[k] * rs;
    attn_h[(size_t)row * L_ + l] = (_Float16)a;
    offset_out[((size_t)b << 20) + (size_t)l * L_ + p] = a;
  }
}

// ---------------------------------------------------------------------------
// Transposed-conv scatter with analytic overlap normalization
// ---------------------------------------------------------------------------
__global__ void ca_assemble(const float* __restrict__ op,
                            float* __restrict__ y0, int total) {
  int idx = blockIdx.x * blockDim.x + threadIdx.x;
  if (idx >= total) return;
  int X = idx & 63, Y = (idx >> 6) & 63, c = (idx >> 12) & 127, b = idx >> 19;
  float s = 0.f;
  int cnt = 0;
#pragma unroll
  for (int i = 0; i < 4; ++i) {
    int ty = Y + 1 - i;
    if (ty < 0 || (ty & 1)) continue;
    int ly = ty >> 1;
    if (ly >= HS) continue;
#pragma unroll
    for (int j = 0; j < 4; ++j) {
      int tx = X + 1 - j;
      if (tx < 0 || (tx & 1)) continue;
      int lx = tx >> 1;
      if (lx >= HS) continue;
      s += op[((size_t)b * L_ + (ly * HS + lx)) * NRAW + (c * 16 + i * 4 + j)];
      ++cnt;
    }
  }
  y0[idx] = s / (float)(cnt ? cnt : 1);
}

// ---------------------------------------------------------------------------
// im2col for 3x3 pad-1 conv on (128,64,64) f32 -> patches f16 [4096,1152]
// ---------------------------------------------------------------------------
__global__ void ca_im2col(const float* __restrict__ in,
                          _Float16* __restrict__ patch, int total) {
  int idx = blockIdx.x * blockDim.x + threadIdx.x;
  if (idx >= total) return;
  int r = idx % KC;
  int pix = idx / KC;
  int cin = r / 9, s = r % 9, di = s / 3, dj = s % 3;
  int y = pix >> 6, x = pix & 63;
  int Y = y + di - 1, X = x + dj - 1;
  float v = 0.f;
  if (Y >= 0 && Y < 64 && X >= 0 && X < 64)
    v = in[((size_t)cin << 12) + (Y << 6) + X];
  patch[idx] = (_Float16)v;
}

__global__ void ca_f32_to_f16(const float* __restrict__ in,
                              _Float16* __restrict__ out, int n) {
  int idx = blockIdx.x * blockDim.x + threadIdx.x;
  if (idx < n) out[idx] = (_Float16)in[idx];
}

// ---------------------------------------------------------------------------
extern "C" void kernel_launch(void* const* d_in, const int* in_sizes, int n_in,
                              void* d_out, int out_size, void* d_ws,
                              size_t ws_size, hipStream_t stream) {
  (void)in_sizes; (void)n_in; (void)out_size; (void)ws_size;
  const float* x    = (const float*)d_in[0];
  const float* mask = (const float*)d_in[1];
  const float* w1   = (const float*)d_in[2];
  const float* b1   = (const float*)d_in[3];
  const float* w2   = (const float*)d_in[4];
  const float* b2   = (const float*)d_in[5];
  float* out = (float*)d_out;
  char* ws = (char*)d_ws;

  size_t off = 0;
  auto alloc = [&](size_t bytes) {
    size_t o = off;
    off = (off + bytes + 255) & ~(size_t)255;
    return o;
  };
  const size_t o_corr  = alloc((size_t)B_ * L_ * L_ * 4);      // 67 MB f32 [b,p,l]
  const size_t o_attnh = alloc((size_t)B_ * L_ * L_ * 2);      // 33.5 MB f16
  const size_t o_rawT  = alloc((size_t)B_ * NRAW * L_ * 2);    // 67 MB f16
  const size_t o_inval = alloc((size_t)B_ * L_ * 4);
  const size_t o_y0    = alloc((size_t)B_ * C_ * H_ * W_ * 4); // 33.5 MB
  const size_t o_y1    = alloc((size_t)B_ * C_ * H_ * W_ * 4);
  const size_t o_patch = alloc((size_t)4096 * KC * 2);         // per-batch reuse
  const size_t o_w1h   = alloc((size_t)C_ * KC * 2);
  const size_t o_w2h   = alloc((size_t)C_ * KC * 2);
  // Union region: {tmp | A | B} lifetime ends before out_patch is produced.
  const size_t o_un    = off;
  const size_t o_tmp   = o_un;                                   // 67 MB f32
  const size_t o_A     = o_un + (size_t)B_ * L_ * L_ * 4;        // 37.7 MB f16
  const size_t o_B     = o_A + (size_t)B_ * L_ * KC * 2;         // 37.7 MB f16
  const size_t o_outp  = o_un;                                   // 134 MB f32

  _Float16* Acols  = (_Float16*)(ws + o_A);
  _Float16* Bcols  = (_Float16*)(ws + o_B);
  _Float16* rawT   = (_Float16*)(ws + o_rawT);
  _Float16* attnh  = (_Float16*)(ws + o_attnh);
  _Float16* patch  = (_Float16*)(ws + o_patch);
  _Float16* w1h    = (_Float16*)(ws + o_w1h);
  _Float16* w2h    = (_Float16*)(ws + o_w2h);
  float* corr  = (float*)(ws + o_corr);
  float* tmp   = (float*)(ws + o_tmp);
  float* inval = (float*)(ws + o_inval);
  float* outp  = (float*)(ws + o_outp);
  float* y0    = (float*)(ws + o_y0);
  float* y1    = (float*)(ws + o_y1);
  float* offset_out = out + (size_t)B_ * C_ * H_ * W_;   // second tuple element

  // 1) patches (A raw / B normalized) + invalid flags
  ca_prep_cols<<<dim3(L_, B_), 128, 0, stream>>>(x, mask, Acols, Bcols, inval);
  // 2) raw 4x4 patches, transposed for deconv GEMM B operand
  {
    int total = B_ * NRAW * L_;
    ca_prep_rawT<<<(total + 255) / 256, 256, 0, stream>>>(x, rawT, total);
  }
  // 3) correlation GEMM: corr[b,p,l]  (M=1024,N=1024,K=1152)
  ca_gemm_staged<<<dim3(L_ / 64, L_ / 128, B_), 256, 0, stream>>>(
      Acols, Bcols, corr, L_, KC,
      (size_t)L_ * KC, (size_t)L_ * KC, (size_t)L_ * L_, nullptr, 0);
  // 4+5) fuse passes + mask + scale
  {
    int total = B_ * L_ * L_;
    ca_fuse1<<<(total + 255) / 256, 256, 0, stream>>>(corr, tmp, total);
    ca_fuse2_mask<<<(total + 255) / 256, 256, 0, stream>>>(tmp, inval, corr, total);
  }
  // 6) softmax -> attn f16 [b,p,l] and f32 offset output [b,l,y,x]
  ca_softmax<<<B_ * L_, 256, 0, stream>>>(corr, attnh, offset_out);
  // 7) deconv GEMM: out_patch[b,p,n]  (M=1024,N=2048,K=1024)
  ca_gemm_staged<<<dim3(NRAW / 64, L_ / 128, B_), 256, 0, stream>>>(
      attnh, rawT, outp, NRAW, L_,
      (size_t)L_ * L_, (size_t)NRAW * L_, (size_t)L_ * NRAW, nullptr, 0);
  // 8) overlap-normalized scatter -> y0 [b,c,Y,X]
  {
    int total = B_ * C_ * H_ * W_;
    ca_assemble<<<(total + 255) / 256, 256, 0, stream>>>(outp, y0, total);
  }
  // 9) conv weights to f16 ([cout, cin*9] == natural flat layout)
  ca_f32_to_f16<<<(C_ * KC + 255) / 256, 256, 0, stream>>>(w1, w1h, C_ * KC);
  ca_f32_to_f16<<<(C_ * KC + 255) / 256, 256, 0, stream>>>(w2, w2h, C_ * KC);
  // 10) two 3x3 conv + ELU via im2col WMMA GEMM, batch-sequential buffer reuse
  const int im_total = 4096 * KC;
  for (int b = 0; b < B_; ++b) {
    const size_t boff = (size_t)b * C_ * H_ * W_;
    ca_im2col<<<(im_total + 255) / 256, 256, 0, stream>>>(y0 + boff, patch, im_total);
    ca_gemm_staged<<<dim3(C_ / 64, 4096 / 128, 1), 256, 0, stream>>>(
        patch, w1h, y1 + boff, C_, KC, 0, 0, 0, b1, 1);
    ca_im2col<<<(im_total + 255) / 256, 256, 0, stream>>>(y1 + boff, patch, im_total);
    ca_gemm_staged<<<dim3(C_ / 64, 4096 / 128, 1), 256, 0, stream>>>(
        patch, w2h, out + boff, C_, KC, 0, 0, 0, b2, 1);
  }
}
